// BiResNet_58548994179648
// MI455X (gfx1250) — compile-verified, matrix operand order checked by reading, not run on hardware
//
#include <hip/hip_runtime.h>
#include <hip/hip_bf16.h>

// ---------------------------------------------------------------------------
// BiResNet forward on MI455X (gfx1250), wave32 + WMMA f32_16x16x32_f16.
// Implicit-GEMM convs and non-local GEMMs run on the matrix pipe with fp32
// accumulation. Activations staged f16 NHWC so each lane's WMMA fragment is
// two contiguous 16B loads. Weights pre-packed in exact B-fragment lane order.
// Round-2 optimizations: A-fragment reuse across all N-tiles in the conv
// kernel (4 accumulators/wave), both P-tiles per wave in NL GEMM2 (single
// pass over the 67MB s16), 8 j-tiles per wave in NL GEMM1, and a fully
// parallel online-softmax (partial max/sumexp -> combine -> normalize)
// instead of 32 serial blocks. ~335MB of d_ws; the 134MB score matrix region
// fits in the 192MB global L2.
// ---------------------------------------------------------------------------

typedef __attribute__((ext_vector_type(16))) _Float16 v16h;
typedef __attribute__((ext_vector_type(8)))  float    v8f;

#define DEV __device__ __forceinline__

DEV v8f vzero8() { return (v8f){0.f, 0.f, 0.f, 0.f, 0.f, 0.f, 0.f, 0.f}; }

// Load A/B fragment: lane m holds K = {lh..lh+7, 16+lh..23+lh} from a row with
// contiguous K (f16). lh = 0 for lanes 0-15, 8 for lanes 16-31.
DEV v16h ld_frag(const _Float16* base, int lh, bool valid) {
  union { v16h v; uint4 q[2]; } f;
  if (valid) {
    f.q[0] = *reinterpret_cast<const uint4*>(base + lh);
    f.q[1] = *reinterpret_cast<const uint4*>(base + lh + 16);
  } else {
    f.q[0] = make_uint4(0u, 0u, 0u, 0u);
    f.q[1] = make_uint4(0u, 0u, 0u, 0u);
  }
  return f.v;
}

// Packed-weight fragment: 16 contiguous halves already in lane order.
DEV v16h ld_frag16(const _Float16* p) {
  union { v16h v; uint4 q[2]; } f;
  f.q[0] = *reinterpret_cast<const uint4*>(p);
  f.q[1] = *reinterpret_cast<const uint4*>(p + 8);
  return f.v;
}

// ---------------------------------------------------------------------------
// conv0: 1->64, 3x3 valid, 130->128, + bias.  Tiny compute (0.6 GFLOP), direct.
// ---------------------------------------------------------------------------
__global__ void k_conv0(const float* __restrict__ x, const float* __restrict__ w,
                        const float* __restrict__ bias, float* __restrict__ out) {
  int idx = blockIdx.x * blockDim.x + threadIdx.x;
  if (idx >= 32 * 64 * 128 * 128) return;
  int xx = idx & 127;
  int yy = (idx >> 7) & 127;
  int co = (idx >> 14) & 63;
  int b  = idx >> 20;
  const float* src = x + (size_t)b * 130 * 130;
  float s = bias[co];
  const float* wc = w + co * 9;
#pragma unroll
  for (int ky = 0; ky < 3; ++ky)
#pragma unroll
    for (int kx = 0; kx < 3; ++kx)
      s += src[(size_t)(yy + ky) * 130 + (xx + kx)] * wc[ky * 3 + kx];
  out[idx] = s;
}

// ---------------------------------------------------------------------------
// Weight packing: W f32 [Co,Ci,k,k] -> pack f16 [taps][Kt][Nt][32 lanes][16]
// in exact B-fragment order (lane n holds column co=nt*16+n, halves are K).
// ---------------------------------------------------------------------------
__global__ void k_pack_w(const float* __restrict__ w, _Float16* __restrict__ pack,
                         int Co, int Ci, int kk, int total) {
  int idx = blockIdx.x * blockDim.x + threadIdx.x;
  if (idx >= total) return;
  int h    = idx & 15;
  int lane = (idx >> 4) & 31;
  int r    = idx >> 9;
  int Nt = Co >> 4, Kt = Ci >> 5;
  int nt = r % Nt; r /= Nt;
  int kt = r % Kt;
  int t  = r / Kt;
  int n  = lane & 15;
  int lh = (lane < 16) ? 0 : 8;
  int k  = (h < 8) ? (lh + h) : (16 + lh + (h - 8));
  int co = (nt << 4) + n;
  int ci = (kt << 5) + k;
  pack[idx] = (_Float16)w[((size_t)co * Ci + ci) * (kk * kk) + t];
}

// ---------------------------------------------------------------------------
// GroupNorm(32 groups of 2 ch) + ReLU.  One block per (b,group).
// ---------------------------------------------------------------------------
__global__ void k_gn_relu(const float* __restrict__ x,
                          const float* __restrict__ gamma, const float* __restrict__ beta,
                          float* of, _Float16* oh, int H, int W) {
  const int C = 64, cpg = 2;
  int bg = blockIdx.x;
  int gi = bg & 31;
  int bb = bg >> 5;
  int c0 = gi * cpg;
  int n  = cpg * H * W;
  const float* base = x + (size_t)(bb * C + c0) * H * W;
  float s = 0.f, s2 = 0.f;
  for (int i = threadIdx.x; i < n; i += 256) { float v = base[i]; s += v; s2 += v * v; }
  __shared__ float rs[256], rq[256];
  rs[threadIdx.x] = s; rq[threadIdx.x] = s2;
  __syncthreads();
  for (int o = 128; o > 0; o >>= 1) {
    if ((int)threadIdx.x < o) { rs[threadIdx.x] += rs[threadIdx.x + o]; rq[threadIdx.x] += rq[threadIdx.x + o]; }
    __syncthreads();
  }
  float mu  = rs[0] / n;
  float var = rq[0] / n - mu * mu;
  float inv = rsqrtf(var + 1e-5f);
  for (int i = threadIdx.x; i < n; i += 256) {
    int c  = c0 + i / (H * W);
    int sp = i % (H * W);
    float v = (base[i] - mu) * inv * gamma[c] + beta[c];
    v = fmaxf(v, 0.f);
    if (of) of[(size_t)(bb * C + c) * H * W + sp] = v;
    if (oh) {
      int y = sp / W, xx = sp % W;
      oh[((size_t)(bb * H + y) * W + xx) * C + c] = (_Float16)v;
    }
  }
}

// f32 NCHW [32,64,32,32] -> f16 NHWC
__global__ void k_to_nhwc(const float* __restrict__ x, _Float16* __restrict__ oh) {
  int idx = blockIdx.x * blockDim.x + threadIdx.x;
  if (idx >= 32 * 64 * 1024) return;
  int sp = idx & 1023;
  int c  = (idx >> 10) & 63;
  int b  = idx >> 16;
  oh[((size_t)b * 1024 + sp) * 64 + c] = (_Float16)x[idx];
}

// ---------------------------------------------------------------------------
// Generic implicit-GEMM conv via WMMA. One wave computes 16 pixels x ALL Co
// channels (Nt<=4 accumulators), reusing each activation A-fragment across
// all N-tiles. Fused residual add (f32 NCHW) + ReLU; emits f32 NCHW and/or
// f16 NHWC and/or f16 NCHW.
// ---------------------------------------------------------------------------
__global__ void k_conv_wmma(const _Float16* __restrict__ inh,
                            const _Float16* __restrict__ pack,
                            float* of, _Float16* ohn, _Float16* ohc,
                            const float* addsrc, int relu,
                            int Bn, int Cin, int Co,
                            int Hin, int Win, int Hout, int Wout,
                            int kh, int stride, int pad) {
  const int lane = threadIdx.x & 31;
  const int Kt = Cin >> 5, Nt = Co >> 4;   // Nt in {2,4}
  const int tilesX = Wout >> 4;
  int wid = blockIdx.x * (blockDim.x >> 5) + (threadIdx.x >> 5);
  int total = Bn * Hout * tilesX;
  if (wid >= total) return;
  int bx = wid % tilesX; int r = wid / tilesX;
  int y  = r % Hout;
  int b  = r / Hout;
  int x0 = bx << 4;
  int m  = lane & 15;
  int lh = (lane < 16) ? 0 : 8;
  v8f acc[4];
  acc[0] = vzero8(); acc[1] = vzero8(); acc[2] = vzero8(); acc[3] = vzero8();
  int taps = kh * kh;
  for (int t = 0; t < taps; ++t) {
    int ky = t / kh, kx = t % kh;
    int iy = y * stride + ky - pad;          // wave-uniform
    if (iy < 0 || iy >= Hin) continue;
    int ix = (x0 + m) * stride + kx - pad;   // per-lane (M row)
    bool vld = (ix >= 0 && ix < Win);
    const _Float16* arow = inh + ((size_t)(b * Hin + iy) * Win + ix) * Cin;
    for (int kt = 0; kt < Kt; ++kt) {
      v16h af = ld_frag(arow + kt * 32, lh, vld);
      const _Float16* bb = pack + ((size_t)(t * Kt + kt) * Nt) * 512 + lane * 16;
      for (int nt = 0; nt < Nt; ++nt) {
        v16h bf = ld_frag16(bb + nt * 512);
        acc[nt] = __builtin_amdgcn_wmma_f32_16x16x32_f16(false, af, false, bf,
                                                         (short)0, acc[nt], false, false);
      }
    }
  }
  int n = lane & 15;
  for (int nt = 0; nt < Nt; ++nt) {
    int co = (nt << 4) + n;
#pragma unroll
    for (int rr = 0; rr < 8; ++rr) {
      int mr = rr + ((lane < 16) ? 0 : 8);   // C/D: VGPR rr -> M=rr (+8 hi lanes)
      int xo = x0 + mr;
      float v = acc[nt][rr];
      size_t onchw = ((size_t)(b * Co + co) * Hout + y) * Wout + xo;
      if (addsrc) v += addsrc[onchw];
      if (relu) v = fmaxf(v, 0.f);
      if (of)  of[onchw] = v;
      if (ohn) ohn[((size_t)(b * Hout + y) * Wout + xo) * Co + co] = (_Float16)v;
      if (ohc) ohc[onchw] = (_Float16)v;
    }
  }
}

// ---------------------------------------------------------------------------
// NL GEMM1: tp[b,i,j] = sum_p ph[b,p,i]*th[b,p,j].  K=P=32 (1 WMMA per tile).
// One wave handles one i-tile x 8 consecutive j-tiles, reusing the A fragment.
// ---------------------------------------------------------------------------
__global__ void k_nl_gemm1(const _Float16* __restrict__ phN,
                           const _Float16* __restrict__ thN,
                           float* __restrict__ tp) {
  const int lane = threadIdx.x & 31;
  int wid = blockIdx.x * (blockDim.x >> 5) + (threadIdx.x >> 5);
  if (wid >= 32 * 64 * 8) return;
  int jt0 = (wid & 7) << 3;
  int it  = (wid >> 3) & 63;
  int b   = wid >> 9;
  int m   = lane & 15;
  int lh  = (lane < 16) ? 0 : 8;
  int n   = lane & 15;
  v16h af = ld_frag(phN + ((size_t)b * 1024 + it * 16 + m) * 32, lh, true);
  for (int j = 0; j < 8; ++j) {
    int jt = jt0 + j;
    v16h bf = ld_frag(thN + ((size_t)b * 1024 + jt * 16 + m) * 32, lh, true);
    v8f acc = vzero8();
    acc = __builtin_amdgcn_wmma_f32_16x16x32_f16(false, af, false, bf,
                                                 (short)0, acc, false, false);
#pragma unroll
    for (int rr = 0; rr < 8; ++rr) {
      int mr = rr + ((lane < 16) ? 0 : 8);
      tp[((size_t)b << 20) + (size_t)(it * 16 + mr) * 1024 + (jt * 16 + n)] = acc[rr];
    }
  }
}

// ---------------------------------------------------------------------------
// Online softmax over the FULL 1M-element flattened score vector per batch
// (reference quirk). 3 fully-parallel kernels:
//  part:   8192 blocks emit (local max m_i, local sum_i = sum exp(v - m_i))
//  comb:   per batch merge: M = max m_i ; S = sum s_i * exp(m_i - M)
//  norm:   s16 = exp(v - M)/S  (f16)
// ---------------------------------------------------------------------------
__global__ void k_sm_part(const float* __restrict__ tp,
                          float* __restrict__ pm, float* __restrict__ ps) {
  int bid = blockIdx.x;            // 32*256 blocks
  int b   = bid >> 8;
  int blk = bid & 255;
  const float* p = tp + ((size_t)b << 20) + (size_t)blk * 4096;
  __shared__ float red[256];
  float mx = -1e30f;
#pragma unroll
  for (int i = 0; i < 16; ++i) mx = fmaxf(mx, p[threadIdx.x + i * 256]);
  red[threadIdx.x] = mx;
  __syncthreads();
  for (int o = 128; o > 0; o >>= 1) {
    if ((int)threadIdx.x < o) red[threadIdx.x] = fmaxf(red[threadIdx.x], red[threadIdx.x + o]);
    __syncthreads();
  }
  mx = red[0];
  __syncthreads();
  float sm = 0.f;
#pragma unroll
  for (int i = 0; i < 16; ++i) sm += __expf(p[threadIdx.x + i * 256] - mx);
  red[threadIdx.x] = sm;
  __syncthreads();
  for (int o = 128; o > 0; o >>= 1) {
    if ((int)threadIdx.x < o) red[threadIdx.x] += red[threadIdx.x + o];
    __syncthreads();
  }
  if (threadIdx.x == 0) { pm[bid] = mx; ps[bid] = red[0]; }
}

__global__ void k_sm_comb(const float* __restrict__ pm, const float* __restrict__ ps,
                          float* __restrict__ Mb, float* __restrict__ Sb) {
  int b = blockIdx.x;              // 32 blocks, 256 threads
  __shared__ float red[256];
  float m = pm[b * 256 + threadIdx.x];
  red[threadIdx.x] = m;
  __syncthreads();
  for (int o = 128; o > 0; o >>= 1) {
    if ((int)threadIdx.x < o) red[threadIdx.x] = fmaxf(red[threadIdx.x], red[threadIdx.x + o]);
    __syncthreads();
  }
  float M = red[0];
  __syncthreads();
  red[threadIdx.x] = ps[b * 256 + threadIdx.x] * __expf(m - M);
  __syncthreads();
  for (int o = 128; o > 0; o >>= 1) {
    if ((int)threadIdx.x < o) red[threadIdx.x] += red[threadIdx.x + o];
    __syncthreads();
  }
  if (threadIdx.x == 0) { Mb[b] = M; Sb[b] = red[0]; }
}

__global__ void k_sm_norm(const float* __restrict__ tp, _Float16* __restrict__ s16,
                          const float* __restrict__ Mb, const float* __restrict__ Sb) {
  size_t idx = (size_t)blockIdx.x * blockDim.x + threadIdx.x;
  if (idx >= ((size_t)32 << 20)) return;
  int b = (int)(idx >> 20);
  s16[idx] = (_Float16)(__expf(tp[idx] - Mb[b]) / Sb[b]);
}

// ---------------------------------------------------------------------------
// NL GEMM2: out[b,p,a] = sum_j s[b,a,j]*g[b,p,j].  K=1024 (32 chunks).
// One wave computes one a-tile x BOTH p-tiles (N=32), single pass over s16.
// Stores y with the reference's per-channel spatial transpose, as f16 NHWC
// [b, hw, 32] for the final 1x1 conv.
// ---------------------------------------------------------------------------
__global__ void k_nl_gemm2(const _Float16* __restrict__ s16,
                           const _Float16* __restrict__ gC,
                           _Float16* __restrict__ y16) {
  const int lane = threadIdx.x & 31;
  int wid = blockIdx.x * (blockDim.x >> 5) + (threadIdx.x >> 5);
  if (wid >= 32 * 64) return;
  int at = wid & 63;
  int b  = wid >> 6;
  int m  = lane & 15;
  int lh = (lane < 16) ? 0 : 8;
  v8f a0 = vzero8(), a1 = vzero8();
  for (int k = 0; k < 32; ++k) {
    const _Float16* arow = s16 + ((size_t)b << 20) + (size_t)(at * 16 + m) * 1024 + k * 32;
    v16h af = ld_frag(arow, lh, true);
    const _Float16* b0 = gC + ((size_t)(b * 32) + m) * 1024 + k * 32;
    const _Float16* b1 = gC + ((size_t)(b * 32) + 16 + m) * 1024 + k * 32;
    v16h bf0 = ld_frag(b0, lh, true);
    v16h bf1 = ld_frag(b1, lh, true);
    a0 = __builtin_amdgcn_wmma_f32_16x16x32_f16(false, af, false, bf0, (short)0, a0, false, false);
    a1 = __builtin_amdgcn_wmma_f32_16x16x32_f16(false, af, false, bf1, (short)0, a1, false, false);
  }
  int n = lane & 15;
#pragma unroll
  for (int rr = 0; rr < 8; ++rr) {
    int a  = at * 16 + rr + ((lane < 16) ? 0 : 8);
    int sp = (a & 31) * 32 + (a >> 5);   // transpose(0,1,3,2) of [B,P,H,H]
    y16[((size_t)b * 1024 + sp) * 32 + n]      = (_Float16)a0[rr];
    y16[((size_t)b * 1024 + sp) * 32 + 16 + n] = (_Float16)a1[rr];
  }
}

// Global average pool (32x32) + linear 64->4.  One block per batch, 64 threads.
__global__ void k_pool_linear(const float* __restrict__ x, const float* __restrict__ w,
                              const float* __restrict__ bias, float* __restrict__ out) {
  int b = blockIdx.x;
  int c = threadIdx.x;   // blockDim == 64
  __shared__ float mean[64];
  const float* p = x + (size_t)(b * 64 + c) * 1024;
  float s = 0.f;
  for (int i = 0; i < 1024; ++i) s += p[i];
  mean[c] = s * (1.f / 1024.f);
  __syncthreads();
  if (c < 4) {
    float o = bias[c];
    for (int k = 0; k < 64; ++k) o += mean[k] * w[c * 64 + k];
    out[b * 4 + c] = o;
  }
}

// ---------------------------------------------------------------------------
// Host orchestration
// ---------------------------------------------------------------------------
extern "C" void kernel_launch(void* const* d_in, const int* in_sizes, int n_in,
                              void* d_out, int out_size, void* d_ws, size_t ws_size,
                              hipStream_t stream) {
  (void)in_sizes; (void)n_in; (void)out_size; (void)ws_size;
  int ii = 0;
  auto IN = [&](void) { return (const float*)d_in[ii++]; };
  const float* stft = IN();
  const float* w0 = IN();
  const float* b0 = IN();
  struct RP { const float *g1, *b1, *w1, *g2, *b2, *w2, *dw; } rp[9];
  for (int r = 0; r < 9; ++r) {
    rp[r].g1 = IN(); rp[r].b1 = IN(); rp[r].w1 = IN();
    rp[r].g2 = IN(); rp[r].b2 = IN(); rp[r].w2 = IN();
    rp[r].dw = (r < 2) ? IN() : nullptr;
  }
  struct NP { const float *th, *ph, *g, *fi; } np[4];
  for (int j = 0; j < 4; ++j) { np[j].th = IN(); np[j].ph = IN(); np[j].g = IN(); np[j].fi = IN(); }
  const float* n0g = IN();
  const float* n0b = IN();
  const float* lw  = IN();
  const float* lb  = IN();

  // ---- workspace carve (~335 MB) ----
  char* ws = (char*)d_ws;
  size_t o = 0;
  auto al = [&](size_t n) { size_t r = o; o += (n + 255) & ~(size_t)255; return ws + r; };
  float*     big1f = (float*)al(134217728);      // conv0 out [32,64,128,128] / NL tp
  _Float16*  big2h = (_Float16*)al(67108864);    // f16 NHWC @128 / NL s16
  float*     f64a  = (float*)al(33554432);       // f32 [32,64,64,64]
  float*     f64b  = (float*)al(33554432);
  _Float16*  h64   = (_Float16*)al(16777216);    // f16 NHWC @64
  float*     xa    = (float*)al(8388608);        // f32 [32,64,32,32]
  float*     xb    = (float*)al(8388608);
  float*     sc32  = (float*)al(8388608);
  _Float16*  h32a  = (_Float16*)al(4194304);     // f16 NHWC @32
  _Float16*  h32b  = (_Float16*)al(4194304);
  _Float16*  thN   = (_Float16*)al(2097152);     // [b,hw,32]
  _Float16*  phN   = (_Float16*)al(2097152);
  _Float16*  gC    = (_Float16*)al(2097152);     // [b,32,hw]
  _Float16*  y16   = (_Float16*)al(2097152);     // [b,hw,32]
  float*     pm    = (float*)al(32 * 256 * 4);   // softmax partial max
  float*     ps    = (float*)al(32 * 256 * 4);   // softmax partial sumexp
  float*     Mb    = (float*)al(32 * 4);
  float*     Sb    = (float*)al(32 * 4);
  _Float16*  cursor = (_Float16*)al(4194304);    // packed weights

  auto packW = [&](const float* w, int Co, int Ci, int k) -> _Float16* {
    int tot = (k * k) * (Ci / 32) * (Co / 16) * 512;
    _Float16* dst = cursor;
    cursor += tot;
    k_pack_w<<<(tot + 255) / 256, 256, 0, stream>>>(w, dst, Co, Ci, k, tot);
    return dst;
  };

  _Float16 *pw1[9], *pw2[9], *pdw[2], *pth[4], *pph[4], *pg[4], *pfi[4];
  for (int r = 0; r < 9; ++r) {
    pw1[r] = packW(rp[r].w1, 64, 64, 3);
    pw2[r] = packW(rp[r].w2, 64, 64, 3);
    if (r < 2) pdw[r] = packW(rp[r].dw, 64, 64, 1);
  }
  for (int j = 0; j < 4; ++j) {
    pth[j] = packW(np[j].th, 32, 64, 1);
    pph[j] = packW(np[j].ph, 32, 64, 1);
    pg[j]  = packW(np[j].g,  32, 64, 1);
    pfi[j] = packW(np[j].fi, 64, 32, 1);
  }

  auto conv = [&](const _Float16* inh, const _Float16* pk,
                  float* of, _Float16* ohn, _Float16* ohc, const float* add, int relu,
                  int Cin, int Co, int Hin, int Hout, int k, int s, int p) {
    int total = 32 * Hout * (Hout / 16);   // one wave covers all Co tiles
    k_conv_wmma<<<(total + 7) / 8, 256, 0, stream>>>(inh, pk, of, ohn, ohc, add, relu,
                                                     32, Cin, Co, Hin, Hin, Hout, Hout, k, s, p);
  };
  auto gn = [&](const float* x, const float* g, const float* b,
                float* of, _Float16* oh, int H) {
    k_gn_relu<<<32 * 32, 256, 0, stream>>>(x, g, b, of, oh, H, H);
  };

  // conv0
  k_conv0<<<(32 * 64 * 128 * 128) / 256, 256, 0, stream>>>(stft, w0, b0, big1f);

  // res00 (stride 2, 128 -> 64)
  gn(big1f, rp[0].g1, rp[0].b1, nullptr, big2h, 128);
  conv(big2h, pdw[0], f64b, nullptr, nullptr, nullptr, 0, 64, 64, 128, 64, 1, 2, 0);
  conv(big2h, pw1[0], f64a, nullptr, nullptr, nullptr, 0, 64, 64, 128, 64, 3, 2, 1);
  gn(f64a, rp[0].g2, rp[0].b2, nullptr, h64, 64);
  conv(h64, pw2[0], f64a, nullptr, nullptr, f64b, 0, 64, 64, 64, 64, 3, 1, 1);

  // res01 (stride 2, 64 -> 32)
  gn(f64a, rp[1].g1, rp[1].b1, nullptr, h64, 64);
  conv(h64, pdw[1], sc32, nullptr, nullptr, nullptr, 0, 64, 64, 64, 32, 1, 2, 0);
  conv(h64, pw1[1], xa, nullptr, nullptr, nullptr, 0, 64, 64, 64, 32, 3, 2, 1);
  gn(xa, rp[1].g2, rp[1].b2, nullptr, h32a, 32);
  conv(h32a, pw2[1], xb, nullptr, nullptr, sc32, 0, 64, 64, 32, 32, 3, 1, 1);

  float* x = xb;
  float* alt = xa;

  auto res32 = [&](const RP& p, _Float16* pk1, _Float16* pk2) {
    gn(x, p.g1, p.b1, nullptr, h32a, 32);
    conv(h32a, pk1, sc32, nullptr, nullptr, nullptr, 0, 64, 64, 32, 32, 3, 1, 1);
    gn(sc32, p.g2, p.b2, nullptr, h32b, 32);
    conv(h32b, pk2, alt, nullptr, nullptr, x, 0, 64, 64, 32, 32, 3, 1, 1);
    float* t = x; x = alt; alt = t;
  };
  auto nl = [&](int j) {
    k_to_nhwc<<<(32 * 64 * 1024 + 255) / 256, 256, 0, stream>>>(x, h32a);
    conv(h32a, pth[j], nullptr, thN, nullptr, nullptr, 0, 64, 32, 32, 32, 1, 1, 0);
    conv(h32a, pph[j], nullptr, phN, nullptr, nullptr, 0, 64, 32, 32, 32, 1, 1, 0);
    conv(h32a, pg[j],  nullptr, nullptr, gC, nullptr, 0, 64, 32, 32, 32, 1, 1, 0);
    k_nl_gemm1<<<(32 * 64 * 8 + 7) / 8, 256, 0, stream>>>(phN, thN, big1f);
    k_sm_part<<<32 * 256, 256, 0, stream>>>(big1f, pm, ps);
    k_sm_comb<<<32, 256, 0, stream>>>(pm, ps, Mb, Sb);
    k_sm_norm<<<(32 * 1024 * 1024) / 256, 256, 0, stream>>>(big1f, big2h, Mb, Sb);
    k_nl_gemm2<<<(32 * 64 + 7) / 8, 256, 0, stream>>>(big2h, gC, y16);
    conv(y16, pfi[j], alt, nullptr, nullptr, x, 1, 32, 64, 32, 32, 1, 1, 0);
    float* t = x; x = alt; alt = t;
  };

  res32(rp[2], pw1[2], pw2[2]);   // res0
  nl(0);                          // nl1
  res32(rp[3], pw1[3], pw2[3]);   // res2
  res32(rp[4], pw1[4], pw2[4]);   // res4
  nl(1);                          // nl2
  res32(rp[5], pw1[5], pw2[5]);   // res6
  res32(rp[6], pw1[6], pw2[6]);   // res8
  nl(2);                          // nl3
  res32(rp[7], pw1[7], pw2[7]);   // res10
  nl(3);                          // nl4
  res32(rp[8], pw1[8], pw2[8]);   // res12

  // final GN+ReLU -> pool -> linear
  gn(x, n0g, n0b, alt, nullptr, 32);
  k_pool_linear<<<32, 64, 0, stream>>>(alt, lw, lb, (float*)d_out);
}
